// spatio_conv_layer_2525440770724
// MI455X (gfx1250) — matrix operand order repaired and checked by reading, not compiled, not run on hardware
//
#include <hip/hip_runtime.h>

// Problem constants (from reference)
#define B_  16
#define C_  64
#define T_  64
#define N_  1024
#define KS_ 3

typedef _Float16 v16h __attribute__((ext_vector_type(16)));
typedef _Float16 v8h  __attribute__((ext_vector_type(8)));
typedef _Float16 v4h  __attribute__((ext_vector_type(4)));
typedef float    v8f  __attribute__((ext_vector_type(8)));
typedef unsigned int v4u __attribute__((ext_vector_type(4)));
typedef int      v4i  __attribute__((ext_vector_type(4)));
typedef int      v8i  __attribute__((ext_vector_type(8)));

// LDS x_c tile: [n_local=64][i*3+k = 192] f16, padded to 200 (400B row, 16B aligned)
#define XC_STRIDE 200

// ---- shared memory layout (byte offsets inside one blob) ----
#define SM_A32   0                       // float  [2][64*32]  = 16384 B (TDM dest, f32 A)
#define SM_A16   16384                   // f16    [64*32]     =  4096 B (converted A)
#define SM_B     20480                   // f16    [2][3*64*32]= 24576 B (TDM dest, f16 B)
#define SM_XC    45056                   // f16    [64*200]    = 25600 B (x_c for GEMM2)
#define SM_TOTAL 70656

// ---------------- pre-pass: Lk f32 -> f16 (same [k][n][m] layout) ----------------
__global__ __launch_bounds__(256) void convert_lk(const float4* __restrict__ lk,
                                                  v4h* __restrict__ lk16) {
  int i = blockIdx.x * 256 + threadIdx.x;   // KS*N*N/4 float4 elements
  float4 f = lk[i];
  v4h h = { (_Float16)f.x, (_Float16)f.y, (_Float16)f.z, (_Float16)f.w };
  lk16[i] = h;
}

// ---------------- pre-pass: theta[i][o][k] f32 -> theta16[o][i*3+k] f16 ----------
__global__ __launch_bounds__(256) void convert_theta(const float* __restrict__ th,
                                                     _Float16* __restrict__ th16) {
  int idx = blockIdx.x * 256 + threadIdx.x;   // C*C*KS = 12288
  int i   = idx / (C_ * KS_);
  int rem = idx % (C_ * KS_);
  int o   = rem / KS_;
  int k   = rem % KS_;
  th16[o * (C_ * KS_) + i * KS_ + k] = (_Float16)th[idx];
}

// ---- TDM: 2D tile load (global -> LDS), D# packed per CDNA5 ISA 8.3/8.4 ----
// dsz_code: 0=1B,1=2B,2=4B elements. Dims/strides in element units.
// clang-23 lane: 6-arg builtin (g0 v4u, g1 v8i, g2 v4i, g3 v4i, extra v8i, cpol)
__device__ __forceinline__ void tdm_load_2d(unsigned lds_addr,
                                            unsigned long long gaddr,
                                            unsigned dsz_code,
                                            unsigned tensor_d0, unsigned tensor_d1,
                                            unsigned tile_d0,   unsigned tile_d1,
                                            unsigned long long stride0) {
  v4u g0;
  g0[0] = 1u;                                              // count=1, user mode
  g0[1] = lds_addr;                                        // LDS byte address
  g0[2] = (unsigned)gaddr;                                 // global addr [31:0]
  g0[3] = (unsigned)((gaddr >> 32) & 0x1FFFFFFull) | (2u << 30);  // [56:32] | type=2
  v8i g1;
  g1[0] = (int)(dsz_code << 16);                           // wg_mask=0, data_size
  g1[1] = (int)((tensor_d0 & 0xFFFFu) << 16);              // tensor_dim0[15:0]
  g1[2] = (int)(((tensor_d0 >> 16) & 0xFFFFu) |
                ((tensor_d1 & 0xFFFFu) << 16));            // dim0 hi | dim1 lo
  g1[3] = (int)(((tensor_d1 >> 16) & 0xFFFFu) |
                ((tile_d0 & 0xFFFFu) << 16));              // dim1 hi | tile_dim0
  g1[4] = (int)(tile_d1 & 0xFFFFu);                        // tile_dim1, tile_dim2=0
  g1[5] = (int)(stride0 & 0xFFFFFFFFull);                  // tensor_dim0_stride lo
  g1[6] = (int)((stride0 >> 32) & 0xFFFFull);              // stride0 hi, stride1=0
  g1[7] = 0;
  v4i g2 = {0, 0, 0, 0};
  v4i g3 = {0, 0, 0, 0};
  v8i g4 = {0, 0, 0, 0, 0, 0, 0, 0};                       // unused VADDR4 slot
  __builtin_amdgcn_tensor_load_to_lds(g0, g1, g2, g3, g4, 0);
}

// ---------------- fused STGCN spatial conv ----------------
// grid = (N/64, B*T); block = 256 (8 waves, wave32)
__global__ __launch_bounds__(256) void stgcn_fused(
    const float*    __restrict__ x,      // [B][C][T][N] f32
    const _Float16* __restrict__ lk16,   // [KS][N][N]   f16
    const _Float16* __restrict__ th16,   // [C_out][C_in*KS] f16
    const float*    __restrict__ bias,   // [C]
    float*          __restrict__ out)    // [B][C][T][N] f32
{
  __shared__ __align__(16) unsigned char smem[SM_TOTAL];
  float    (*aBuf32)[64 * 32]       = (float(*)[64 * 32])(smem + SM_A32);
  _Float16* a16                     = (_Float16*)(smem + SM_A16);
  _Float16 (*bBuf)[KS_ * 64 * 32]   = (_Float16(*)[KS_ * 64 * 32])(smem + SM_B);
  _Float16* xcT                     = (_Float16*)(smem + SM_XC);
  const unsigned smBase = (unsigned)(size_t)(void*)smem;   // LDS byte offset (addr[31:0])

  const int bt = blockIdx.y;
  const int b  = bt >> 6;          // /T
  const int t  = bt & 63;          // %T
  const int n0 = blockIdx.x << 6;  // 64-wide n block

  const int wave  = (int)(threadIdx.x >> 5);
  const int lane  = (int)(threadIdx.x & 31);
  const int lhalf = lane >> 4;     // lane half selects K-subchunk
  const int lmod  = lane & 15;     // row/col within 16-wide tile

  // ======== per-wave TDM slice (all 8 waves issue exactly one TDM per K-step;
  //          no control flow around the TDM since it ignores EXEC) ========
  // waves 0-3: A slice = 16 rows of x[b, i, t, :] (f32, row stride T*N)
  // waves 4-7: B slice = Lk16[k][n0+..][:] rows (f16, row stride N); k=2 split in 2
  const bool     isA  = wave < 4;
  const unsigned wa   = (unsigned)(wave & 3);
  const unsigned kb   = isA ? 0u : ((unsigned)(wave - 4) > 2u ? 2u : (unsigned)(wave - 4));
  const unsigned noff = (wave == 7) ? 32u : 0u;
  const unsigned rows = isA ? 16u : ((wave >= 6) ? 32u : 64u);
  const unsigned dszc = isA ? 2u : 1u;     // 4B : 2B elements
  const unsigned esz  = isA ? 4u : 2u;
  const unsigned long long stride0 =
      isA ? (unsigned long long)(T_ * N_) : (unsigned long long)N_;
  const unsigned long long gbaseA =
      (unsigned long long)(size_t)(x + (((size_t)(b * C_ + 16 * wa)) * T_ + t) * N_);
  const unsigned long long gbaseB =
      (unsigned long long)(size_t)(lk16 + ((size_t)kb * N_ + (size_t)(n0 + noff)) * N_);
  const unsigned long long gbase = isA ? gbaseA : gbaseB;
  const unsigned ldsBase = isA ? (smBase + SM_A32 + wa * (16u * 32u * 4u))
                               : (smBase + SM_B + (kb * 64u + noff) * 32u * 2u);
  const unsigned ldsStride = isA ? (64u * 32u * 4u) : (unsigned)(KS_ * 64 * 32 * 2);

  // ============ GEMM1: x_c[i,k,n] = sum_m x[b,i,t,m] * Lk[k][n][m] ============
  const int wm = wave & 1;   // 32-row (i) half
  const int wn = wave >> 1;  // 16-col (n) quarter

  v8f acc[2][KS_];
  for (int mt = 0; mt < 2; ++mt)
    for (int k = 0; k < KS_; ++k)
      for (int r = 0; r < 8; ++r) acc[mt][k][r] = 0.0f;

  const int NSTEP = N_ / 32;

  // prologue: load step 0 into buffer 0
  tdm_load_2d(ldsBase, gbase, dszc, (unsigned)N_, rows, 32u, rows, stride0);

  for (int s = 0; s < NSTEP; ++s) {
    const int cur = s & 1;
    __builtin_amdgcn_s_wait_tensorcnt((short)0);   // this wave's slice landed
    __syncthreads();                               // all slices visible; prev bufs free
    if (s + 1 < NSTEP) {                           // uniform scalar branch
      tdm_load_2d(ldsBase + (unsigned)(1 - cur) * ldsStride,
                  gbase + (unsigned long long)((s + 1) * 32) * esz,
                  dszc, (unsigned)N_, rows, 32u, rows, stride0);
    }
    // cooperative A tile convert: f32[64][32] -> f16[64][32]
    {
      const int flat = (int)threadIdx.x * 8;       // 2048 elements / 256 threads
      const float* sp = &aBuf32[cur][flat];
      float4 f0 = *(const float4*)sp;
      float4 f1 = *(const float4*)(sp + 4);
      v8h h = { (_Float16)f0.x, (_Float16)f0.y, (_Float16)f0.z, (_Float16)f0.w,
                (_Float16)f1.x, (_Float16)f1.y, (_Float16)f1.z, (_Float16)f1.w };
      *(v8h*)&a16[flat] = h;
    }
    __syncthreads();                               // converted A visible

    // fragments from LDS + WMMA
    v16h afrag[2];
    for (int mt = 0; mt < 2; ++mt) {
      const _Float16* ap = &a16[(32 * wm + 16 * mt + lmod) * 32 + 8 * lhalf];
      v8h a0 = *(const v8h*)ap;
      v8h a1 = *(const v8h*)(ap + 16);
      afrag[mt] = __builtin_shufflevector(a0, a1, 0,1,2,3,4,5,6,7,8,9,10,11,12,13,14,15);
    }
    for (int k = 0; k < KS_; ++k) {
      const _Float16* bp = &bBuf[cur][(k * 64 + 16 * wn + lmod) * 32 + 8 * lhalf];
      v8h b0 = *(const v8h*)bp;
      v8h b1 = *(const v8h*)(bp + 16);
      v16h bf = __builtin_shufflevector(b0, b1, 0,1,2,3,4,5,6,7,8,9,10,11,12,13,14,15);
      for (int mt = 0; mt < 2; ++mt) {
        acc[mt][k] = __builtin_amdgcn_wmma_f32_16x16x32_f16(
            false, afrag[mt], false, bf, (short)0, acc[mt][k], false, false);
      }
    }
  }

  // Dump x_c to LDS as f16 in [n_local][i*3+k] layout (B-fragment layout for GEMM2).
  for (int mt = 0; mt < 2; ++mt)
    for (int k = 0; k < KS_; ++k)
      for (int r = 0; r < 8; ++r) {
        int i  = 32 * wm + 16 * mt + r + 8 * lhalf;  // C/D layout: VGPR r -> M=r / r+8
        int nl = 16 * wn + lmod;
        xcT[nl * XC_STRIDE + i * KS_ + k] = (_Float16)acc[mt][k][r];
      }
  __syncthreads();

  // ============ GEMM2: out[o,n] = sum_{ik} theta16[o][ik] * xcT[n][ik] ============
  const int ot = wave & 3;    // o tile (16 wide)
  const int np = wave >> 2;   // n pair (two 16-wide n tiles)
  v8f acc2[2];
  for (int j = 0; j < 2; ++j)
    for (int r = 0; r < 8; ++r) acc2[j][r] = 0.0f;

  const _Float16* throw_ = th16 + (size_t)(16 * ot + lmod) * (C_ * KS_);
  for (int mm0 = 0; mm0 < C_ * KS_; mm0 += 32) {
    const _Float16* p = throw_ + mm0 + 8 * lhalf;
    v8h a0 = *(const v8h*)p;
    v8h a1 = *(const v8h*)(p + 16);
    v16h af = __builtin_shufflevector(a0, a1, 0,1,2,3,4,5,6,7,8,9,10,11,12,13,14,15);
    for (int j = 0; j < 2; ++j) {
      int nl = 32 * np + 16 * j + lmod;
      const _Float16* q = &xcT[nl * XC_STRIDE + mm0 + 8 * lhalf];
      v8h b0 = *(const v8h*)q;
      v8h b1 = *(const v8h*)(q + 16);
      v16h bf = __builtin_shufflevector(b0, b1, 0,1,2,3,4,5,6,7,8,9,10,11,12,13,14,15);
      acc2[j] = __builtin_amdgcn_wmma_f32_16x16x32_f16(
          false, af, false, bf, (short)0, acc2[j], false, false);
    }
  }

  // Epilogue: + bias + residual, ReLU, store f32.
  for (int j = 0; j < 2; ++j)
    for (int r = 0; r < 8; ++r) {
      int o  = 16 * ot + r + 8 * lhalf;
      int nl = 32 * np + 16 * j + lmod;
      int n  = n0 + nl;
      size_t idx = (((size_t)(b * C_ + o)) * T_ + t) * N_ + n;
      float v = acc2[j][r] + bias[o] + x[idx];
      out[idx] = v > 0.0f ? v : 0.0f;
    }
}

// ---------------- launcher ----------------
extern "C" void kernel_launch(void* const* d_in, const int* in_sizes, int n_in,
                              void* d_out, int out_size, void* d_ws, size_t ws_size,
                              hipStream_t stream) {
  (void)in_sizes; (void)n_in; (void)out_size; (void)ws_size;
  const float* x  = (const float*)d_in[0];
  const float* lk = (const float*)d_in[1];
  const float* th = (const float*)d_in[2];
  const float* bs = (const float*)d_in[3];
  float* out = (float*)d_out;

  // workspace: [0, 6 MB) = Lk f16; then 24 KB theta16
  _Float16* lk16 = (_Float16*)d_ws;
  _Float16* th16 = (_Float16*)((char*)d_ws + (size_t)KS_ * N_ * N_ * sizeof(_Float16));

  convert_lk<<<(KS_ * N_ * N_ / 4) / 256, 256, 0, stream>>>((const float4*)lk, (v4h*)lk16);
  convert_theta<<<(C_ * C_ * KS_) / 256, 256, 0, stream>>>(th, th16);

  dim3 grid(N_ / 64, B_ * T_);
  stgcn_fused<<<grid, 256, 0, stream>>>(x, lk16, th16, bs, out);
}
